// MultiHeadAttentionWithRPE_18511309045831
// MI455X (gfx1250) — matrix operand order repaired and checked by reading, not compiled, not run on hardware
//
#include <hip/hip_runtime.h>

// ---------------------------------------------------------------------------
// MultiHeadAttention with relative position embeddings — CDNA5 / gfx1250
// B=2, S=2048, D=768, H=12, dk=64, MAX_REL=32
// - all matmuls on v_wmma_f32_16x16x32_bf16 (wave32)
// - B-side operands pre-swizzled in global memory into the exact WMMA
//   B-fragment register image: fragment = two global_load_b128
// - GEMM A-tiles streamed into LDS by the Tensor Data Mover (tensor_load_to_lds),
//   double-buffered and synchronized with s_wait_tensorcnt
// ---------------------------------------------------------------------------

typedef __bf16 v16bf __attribute__((ext_vector_type(16)));
typedef float  v8f   __attribute__((ext_vector_type(8)));
typedef unsigned int u32x4v __attribute__((ext_vector_type(4)));
typedef int          i32x4v __attribute__((ext_vector_type(4)));
typedef int          i32x8v __attribute__((ext_vector_type(8)));

union BF16x16 {
    v16bf          v;
    unsigned short u[16];
    unsigned int   w[8];
};

__device__ __forceinline__ unsigned short f32_to_bf16(float f) {
    unsigned int x = __float_as_uint(f);
    unsigned int r = (x + 0x7FFFu + ((x >> 16) & 1u)) >> 16;
    return (unsigned short)r;
}

__device__ __forceinline__ unsigned int pack_bf16x2(float lo, float hi) {
    return (unsigned int)f32_to_bf16(lo) | ((unsigned int)f32_to_bf16(hi) << 16);
}

__device__ __forceinline__ v8f wmma_bf16(const BF16x16& a, const BF16x16& b, v8f c) {
    return __builtin_amdgcn_wmma_f32_16x16x32_bf16(false, a.v, false, b.v,
                                                   (short)0, c, false, false);
}

// ---- Tensor Data Mover: 2D tile load Global->LDS (D# per ISA 08 §8) --------
// dims/strides in elements (data_size = 2 bytes), addresses in bytes.
__device__ __forceinline__ void tdm_load_2d(unsigned int lds_off, const void* gptr,
                                            unsigned int tdim0, unsigned int tdim1,
                                            unsigned int tile0, unsigned int tile1,
                                            unsigned int stride0) {
    unsigned long long ga = (unsigned long long)(uintptr_t)gptr;
    union { unsigned int a[4]; u32x4v v; } g0;
    g0.a[0] = 1u;                                    // count=1, user mode
    g0.a[1] = lds_off;                               // lds_addr [63:32]
    g0.a[2] = (unsigned int)ga;                      // global_addr lo
    g0.a[3] = (unsigned int)((ga >> 32) & 0x1FFFFFFu) | (2u << 30); // addr hi | type=2
    union { unsigned int a[8]; i32x8v v; } g1;
    g1.a[0] = 1u << 16;                              // data_size = 2 bytes
    g1.a[1] = (tdim0 & 0xFFFFu) << 16;               // tensor_dim0 [79:48] lo
    g1.a[2] = (tdim0 >> 16) | ((tdim1 & 0xFFFFu) << 16); // dim0 hi | tensor_dim1 lo
    g1.a[3] = (tdim1 >> 16) | (tile0 << 16);         // dim1 hi | tile_dim0 [127:112]
    g1.a[4] = tile1;                                 // tile_dim1 [143:128]
    g1.a[5] = stride0;                               // tensor_dim0_stride [207:160]
    g1.a[6] = 0u;
    g1.a[7] = 0u;
    i32x4v z4 = {0, 0, 0, 0};
#if __has_include(<hip/amd_detail/amd_gfx1250_TDM.h>)
    i32x8v z8 = {0, 0, 0, 0, 0, 0, 0, 0};            // 6-arg toolchain
    __builtin_amdgcn_tensor_load_to_lds(g0.v, g1.v, z4, z4, z8, 0);
#else
    __builtin_amdgcn_tensor_load_to_lds(g0.v, g1.v, z4, z4, 0); // ROCm 7.2, 5-arg
#endif
}

// --------------------------- constants -------------------------------------
#define BATCH 2
#define SEQ   2048
#define DMODEL 768
#define NHEAD 12
#define DK    64
#define BH    (BATCH * NHEAD)
#define PREL  80          // 65 rel rows padded to 80 (5 n-tiles of 16)

// --------------------------- f32 -> bf16 convert ----------------------------
__global__ void cvt_bf16_kernel(const float* __restrict__ src,
                                unsigned short* __restrict__ dst, int n) {
    int i = blockIdx.x * 256 + threadIdx.x;
    if (i < n) dst[i] = f32_to_bf16(src[i]);
}

// W [K=768][N=768] f32 -> swizzled bf16-pair dwords in B-fragment order:
//   out[((kt*N + n)*2 + khalf)*8 + v] = { W[k0][n], W[k0+1][n] },
//   k0 = kt*32 + khalf*16 + 2v
__global__ void wpack_kernel(const float* __restrict__ W,
                             unsigned int* __restrict__ out) {
    int t = blockIdx.x * 256 + threadIdx.x;      // total (768/32)*768*16 = 294912
    if (t >= (DMODEL / 32) * DMODEL * 16) return;
    int v     = t & 7;
    int khalf = (t >> 3) & 1;
    int n     = (t >> 4) % DMODEL;
    int kt    = t / (16 * DMODEL);
    int k0    = kt * 32 + khalf * 16 + 2 * v;
    out[t] = pack_bf16x2(W[(size_t)k0 * DMODEL + n], W[(size_t)(k0 + 1) * DMODEL + n]);
}

// rel_emb [65][64] f32 -> swizzled B-fragment dwords (contraction over d)
__global__ void relpack_kernel(const float* __restrict__ rel,
                               unsigned int* __restrict__ out) {
    int t = blockIdx.x * 256 + threadIdx.x;      // total 2*80*16 = 2560
    if (t >= 2 * PREL * 16) return;
    int v     = t & 7;
    int khalf = (t >> 3) & 1;
    int r     = (t >> 4) % PREL;
    int kt    = t / (16 * PREL);
    int d0    = kt * 32 + khalf * 16 + 2 * v;
    out[t] = (r < 65) ? pack_bf16x2(rel[r * DK + d0], rel[r * DK + d0 + 1]) : 0u;
}

// --------------------------- generic bf16 GEMM ------------------------------
// C[M,N] = A[M,K] * Bswz + bias.  A bf16 row-major, B pre-swizzled (wpack).
// A tiles (128x32) streamed Global->LDS by the TDM, double-buffered.
// MODE 0: dst bf16, head-split [B,H,S,dk]                 (Q,K projections)
// MODE 2: dst f32 row-major                               (final output GEMM)
// MODE 3: dst uint, V in swizzled B-fragment layout       (V projection)
template <int MODE>
__global__ void gemm_bf16_kernel(const unsigned short* __restrict__ A,
                                 const unsigned int* __restrict__ Bswz,
                                 const float* __restrict__ bias,
                                 void* __restrict__ dst,
                                 int M, int N, int K) {
    __shared__ __align__(16) unsigned short At[2][128 * 32];   // double buffer

    const int tid  = threadIdx.x;
    const int lane = tid & 31;
    const int wv   = tid >> 5;
    const int m0   = blockIdx.x * 128;
    const int n0   = blockIdx.y * 64;

    v8f acc[4] = {};

    // prologue: TDM kick for tile kk=0 (wave 0 owns the DMA; TENSORcnt is per-wave)
    if (wv == 0)
        tdm_load_2d((unsigned int)(uintptr_t)&At[0][0],
                    A + (size_t)m0 * K, K, M, 32, 128, K);

    int cur = 0;
    for (int kk = 0; kk < K; kk += 32) {
        if (wv == 0) {
            if (kk + 32 < K) { // stream next tile while this one is consumed
                tdm_load_2d((unsigned int)(uintptr_t)&At[cur ^ 1][0],
                            A + (size_t)m0 * K + kk + 32, K, M, 32, 128, K);
                __builtin_amdgcn_s_wait_tensorcnt(1);
            } else {
                __builtin_amdgcn_s_wait_tensorcnt(0);
            }
        }
        __syncthreads();   // tile At[cur] ready for all waves

        // A fragment: 16x32 contiguous layout (lanes 0-15: K 0-15; 16-31: K 16-31)
        BF16x16 af;
        {
            int row = wv * 16 + (lane & 15);
            int kg  = (lane >> 4) * 16;
            *reinterpret_cast<uint4*>(&af.w[0]) =
                *reinterpret_cast<const uint4*>(&At[cur][row * 32 + kg]);
            *reinterpret_cast<uint4*>(&af.w[4]) =
                *reinterpret_cast<const uint4*>(&At[cur][row * 32 + 8 + kg]);
        }
#pragma unroll
        for (int nt = 0; nt < 4; ++nt) {
            // B fragment: two b128 loads from swizzled weights (L2-resident)
            const unsigned int* bp =
                Bswz + (((size_t)(kk >> 5) * N + n0 + nt * 16 + (lane & 15)) * 2 + (lane >> 4)) * 8;
            BF16x16 bf;
            *reinterpret_cast<uint4*>(&bf.w[0]) = *reinterpret_cast<const uint4*>(bp);
            *reinterpret_cast<uint4*>(&bf.w[4]) = *reinterpret_cast<const uint4*>(bp + 4);
            acc[nt] = wmma_bf16(af, bf, acc[nt]);
        }
        __syncthreads();   // all waves done with At[cur] before TDM overwrites it
        cur ^= 1;
    }

    // ---- epilogue ----
    const int rbase = m0 + wv * 16 + (lane >> 4) * 8;  // 8 consecutive M rows
#pragma unroll
    for (int nt = 0; nt < 4; ++nt) {
        int col = n0 + nt * 16 + (lane & 15);
        float bv = bias ? bias[col] : 0.0f;
        if (MODE == 3) { // V: write swizzled B-fragment layout (key pairs -> dwords)
            int h = col >> 6, d = col & 63;
#pragma unroll
            for (int vp = 0; vp < 4; ++vp) {
                int row0 = rbase + 2 * vp;
                int b = row0 >> 11, s = row0 & 2047;
                unsigned int dw = pack_bf16x2(acc[nt][2 * vp] + bv, acc[nt][2 * vp + 1] + bv);
                ((unsigned int*)dst)[(((size_t)(b * NHEAD + h) * (SEQ / 32) + (s >> 5)) * DK + d) * 16 +
                                     ((s >> 4) & 1) * 8 + ((s & 15) >> 1)] = dw;
            }
        } else {
#pragma unroll
            for (int v = 0; v < 8; ++v) {
                float val = acc[nt][v] + bv;
                int row = rbase + v;
                if (MODE == 0) { // head-split bf16 [B,H,S,dk]
                    int h = col >> 6, d = col & 63;
                    int b = row >> 11, s = row & 2047;
                    ((unsigned short*)dst)[(((size_t)(b * NHEAD + h) * SEQ) + s) * DK + d] =
                        f32_to_bf16(val);
                } else {         // f32 row-major
                    ((float*)dst)[(size_t)row * N + col] = val;
                }
            }
        }
    }
}

// --------------------------- P = Q @ rel^T ----------------------------------
// P[bh, q, r] = sum_d Q[bh,q,d] * rel_emb[r,d]   (r padded to 80, no LDS)
__global__ void pgemm_kernel(const unsigned short* __restrict__ Qbf,    // [BH,S,64]
                             const unsigned int* __restrict__ relswz,   // packed
                             float* __restrict__ P) {                   // [BH,S,80]
    const int tid = threadIdx.x, lane = tid & 31, wv = tid >> 5;
    const int bh = blockIdx.y;
    const int q0 = blockIdx.x * 128;

    const int row = q0 + wv * 16 + (lane & 15);
    const unsigned short* qrow = Qbf + ((size_t)bh * SEQ + row) * DK;

    v8f acc[5] = {};
#pragma unroll
    for (int ks = 0; ks < 2; ++ks) {
        BF16x16 af;
        int kg = (lane >> 4) * 16;
        *reinterpret_cast<uint4*>(&af.w[0]) = *reinterpret_cast<const uint4*>(qrow + ks * 32 + kg);
        *reinterpret_cast<uint4*>(&af.w[4]) = *reinterpret_cast<const uint4*>(qrow + ks * 32 + 8 + kg);
#pragma unroll
        for (int nt = 0; nt < 5; ++nt) {
            const unsigned int* bp =
                relswz + (((size_t)ks * PREL + nt * 16 + (lane & 15)) * 2 + (lane >> 4)) * 8;
            BF16x16 bf;
            *reinterpret_cast<uint4*>(&bf.w[0]) = *reinterpret_cast<const uint4*>(bp);
            *reinterpret_cast<uint4*>(&bf.w[4]) = *reinterpret_cast<const uint4*>(bp + 4);
            acc[nt] = wmma_bf16(af, bf, acc[nt]);
        }
    }

    const int r2 = q0 + wv * 16 + (lane >> 4) * 8;
    float* Pb = P + (size_t)bh * SEQ * PREL;
#pragma unroll
    for (int nt = 0; nt < 5; ++nt)
#pragma unroll
        for (int v = 0; v < 8; ++v)
            Pb[(size_t)(r2 + v) * PREL + nt * 16 + (lane & 15)] = acc[nt][v];
}

// --------------------------- flash attention --------------------------------
// grid (S/128, BH); 8 independent waves, 16 q-rows each; no workgroup barriers.
// K fragments stream from row-major Kbf (pairs contiguous along d);
// V fragments stream from pre-swizzled Vswz; both L2-resident.
__global__ void attn_kernel(const unsigned short* __restrict__ Qbf,
                            const unsigned short* __restrict__ Kbf,
                            const unsigned int* __restrict__ Vswz,
                            const float* __restrict__ P,
                            unsigned short* __restrict__ ctx) { // bf16 [B,S,768]
    __shared__ __align__(16) unsigned short Pt[8][16 * 32];     // probs relayout

    const int tid = threadIdx.x, lane = tid & 31, wv = tid >> 5;
    const int bh = blockIdx.y;
    const int h = bh % NHEAD, b = bh / NHEAD;
    const int q0 = blockIdx.x * 128 + wv * 16;
    const size_t base = (size_t)bh * SEQ * DK;
    const float scale = 0.125f; // 1/sqrt(64)

    // persistent Q fragments: d 0..31 (aq0) and 32..63 (aq1)
    BF16x16 aq0, aq1;
    {
        const unsigned short* qrow = Qbf + base + (size_t)(q0 + (lane & 15)) * DK;
        int kg = (lane >> 4) * 16;
        *reinterpret_cast<uint4*>(&aq0.w[0]) = *reinterpret_cast<const uint4*>(qrow + kg);
        *reinterpret_cast<uint4*>(&aq0.w[4]) = *reinterpret_cast<const uint4*>(qrow + 8 + kg);
        *reinterpret_cast<uint4*>(&aq1.w[0]) = *reinterpret_cast<const uint4*>(qrow + 32 + kg);
        *reinterpret_cast<uint4*>(&aq1.w[4]) = *reinterpret_cast<const uint4*>(qrow + 40 + kg);
    }

    const int rbase = q0 + (lane >> 4) * 8;          // first q-row this lane owns
    const float* Pbh = P + (size_t)bh * SEQ * PREL;
    float m[8], l[8], Plo[8], Phi[8];
#pragma unroll
    for (int v = 0; v < 8; ++v) {
        m[v] = -1e30f; l[v] = 0.0f;
        Plo[v] = Pbh[(size_t)(rbase + v) * PREL + 0];   // idx 0  (q-k <= -32)
        Phi[v] = Pbh[(size_t)(rbase + v) * PREL + 64];  // idx 64 (q-k >= 32)
    }
    v8f acc[4] = {};
    const int khalf16 = (lane >> 4) * 16;

    for (int kb = 0; kb < SEQ; kb += 32) {
        if (kb + 32 < SEQ) { // prefetch next key block rows
            __builtin_prefetch(Kbf + base + (size_t)(kb + 32 + (lane & 15)) * DK, 0, 1);
            __builtin_prefetch(Vswz + (((size_t)bh * (SEQ / 32) + ((kb >> 5) + 1)) * DK) * 16, 0, 1);
        }

        // ---- scores: two 16x16 tiles (keys kb..kb+15, kb+16..kb+31) ----
        v8f s0 = {}, s1 = {};
#pragma unroll
        for (int t = 0; t < 2; ++t) {
            const unsigned short* krow = Kbf + base + (size_t)(kb + t * 16 + (lane & 15)) * DK;
            BF16x16 bk;
            v8f sv = t ? s1 : s0;
            *reinterpret_cast<uint4*>(&bk.w[0]) = *reinterpret_cast<const uint4*>(krow + khalf16);
            *reinterpret_cast<uint4*>(&bk.w[4]) = *reinterpret_cast<const uint4*>(krow + khalf16 + 8);
            sv = wmma_bf16(aq0, bk, sv);
            *reinterpret_cast<uint4*>(&bk.w[0]) = *reinterpret_cast<const uint4*>(krow + 32 + khalf16);
            *reinterpret_cast<uint4*>(&bk.w[4]) = *reinterpret_cast<const uint4*>(krow + 32 + khalf16 + 8);
            sv = wmma_bf16(aq1, bk, sv);
            if (t) s1 = sv; else s0 = sv;
        }

        // ---- relative position bias (clamped fast paths off-diagonal) ----
        if (q0 - (kb + 31) >= 32) {              // all q-k >= 32 -> idx 64
#pragma unroll
            for (int v = 0; v < 8; ++v) { s0[v] += Phi[v]; s1[v] += Phi[v]; }
        } else if ((q0 + 15) - kb <= -32) {      // all q-k <= -32 -> idx 0
#pragma unroll
            for (int v = 0; v < 8; ++v) { s0[v] += Plo[v]; s1[v] += Plo[v]; }
        } else {                                  // near-diagonal: gather
#pragma unroll
            for (int v = 0; v < 8; ++v) {
                int q = rbase + v;
                int d0i = q - (kb + (lane & 15));
                int d1i = d0i - 16;
                int i0 = min(max(d0i, -32), 32) + 32;
                int i1 = min(max(d1i, -32), 32) + 32;
                s0[v] += Pbh[(size_t)q * PREL + i0];
                s1[v] += Pbh[(size_t)q * PREL + i1];
            }
        }

        // ---- online softmax (row reductions across 16-lane half-wave) ----
#pragma unroll
        for (int v = 0; v < 8; ++v) {
            float a0 = s0[v] * scale, a1 = s1[v] * scale;
            float mx = fmaxf(a0, a1);
            mx = fmaxf(mx, __shfl_xor(mx, 1, 32));
            mx = fmaxf(mx, __shfl_xor(mx, 2, 32));
            mx = fmaxf(mx, __shfl_xor(mx, 4, 32));
            mx = fmaxf(mx, __shfl_xor(mx, 8, 32));
            float mn = fmaxf(m[v], mx);
            float al = __expf(m[v] - mn);
            float p0 = __expf(a0 - mn);
            float p1 = __expf(a1 - mn);
            float ls = p0 + p1;
            ls += __shfl_xor(ls, 1, 32);
            ls += __shfl_xor(ls, 2, 32);
            ls += __shfl_xor(ls, 4, 32);
            ls += __shfl_xor(ls, 8, 32);
            l[v] = l[v] * al + ls;
            m[v] = mn;
#pragma unroll
            for (int n = 0; n < 4; ++n) acc[n][v] *= al;
            // stash probs (C layout -> LDS, re-read in A layout)
            int pr = (v + 8 * (lane >> 4)) * 32 + (lane & 15);
            Pt[wv][pr]      = f32_to_bf16(p0);
            Pt[wv][pr + 16] = f32_to_bf16(p1);
        }
        __builtin_amdgcn_wave_barrier();   // per-wave DS ops are in order

        // ---- ctx += probs(16x32) @ V(32x64) ----
        BF16x16 ap;
        {
            int pr = (lane & 15) * 32;
            int kg = (lane >> 4) * 16;
            *reinterpret_cast<uint4*>(&ap.w[0]) = *reinterpret_cast<const uint4*>(&Pt[wv][pr + kg]);
            *reinterpret_cast<uint4*>(&ap.w[4]) = *reinterpret_cast<const uint4*>(&Pt[wv][pr + 8 + kg]);
        }
        const unsigned int* vb = Vswz + (((size_t)bh * (SEQ / 32) + (kb >> 5)) * DK) * 16;
#pragma unroll
        for (int nt = 0; nt < 4; ++nt) {
            const unsigned int* bp = vb + (size_t)(nt * 16 + (lane & 15)) * 16 + (lane >> 4) * 8;
            BF16x16 bv;
            *reinterpret_cast<uint4*>(&bv.w[0]) = *reinterpret_cast<const uint4*>(bp);
            *reinterpret_cast<uint4*>(&bv.w[4]) = *reinterpret_cast<const uint4*>(bp + 4);
            acc[nt] = wmma_bf16(ap, bv, acc[nt]);
        }
    }

    // ---- normalize + write ctx as bf16 [B,S,D] (head-interleaved cols) ----
#pragma unroll
    for (int v = 0; v < 8; ++v) {
        float inv = 1.0f / l[v];
        int row = rbase + v;
        size_t o = ((size_t)b * SEQ + row) * DMODEL + h * DK;
#pragma unroll
        for (int nt = 0; nt < 4; ++nt)
            ctx[o + nt * 16 + (lane & 15)] = f32_to_bf16(acc[nt][v] * inv);
    }
}

// --------------------------- launcher ---------------------------------------
extern "C" void kernel_launch(void* const* d_in, const int* in_sizes, int n_in,
                              void* d_out, int out_size, void* d_ws, size_t ws_size,
                              hipStream_t stream) {
    const float* x   = (const float*)d_in[0];
    const float* Wq  = (const float*)d_in[1];
    const float* bq  = (const float*)d_in[2];
    const float* Wk  = (const float*)d_in[3];
    const float* bk  = (const float*)d_in[4];
    const float* Wv  = (const float*)d_in[5];
    const float* bv  = (const float*)d_in[6];
    const float* Wo  = (const float*)d_in[7];
    const float* bo  = (const float*)d_in[8];
    const float* rel = (const float*)d_in[9];

    char* ws = (char*)d_ws;
    size_t off = 0;
    auto alloc = [&](size_t bytes) -> char* {
        char* p = ws + off;
        off += (bytes + 255) & ~(size_t)255;
        return p;
    };

    const size_t MROWS = (size_t)BATCH * SEQ;                    // 4096
    const size_t WSWZ  = (size_t)(DMODEL / 32) * DMODEL * 16;    // dwords per weight
    unsigned short* xbf    = (unsigned short*)alloc(MROWS * DMODEL * 2);
    unsigned int*   wq_swz = (unsigned int*)alloc(WSWZ * 4);
    unsigned int*   wk_swz = (unsigned int*)alloc(WSWZ * 4);
    unsigned int*   wv_swz = (unsigned int*)alloc(WSWZ * 4);
    unsigned int*   wo_swz = (unsigned int*)alloc(WSWZ * 4);
    unsigned int*   relswz = (unsigned int*)alloc((size_t)2 * PREL * 16 * 4);
    unsigned short* qbf    = (unsigned short*)alloc((size_t)BH * SEQ * DK * 2);
    unsigned short* kbf    = (unsigned short*)alloc((size_t)BH * SEQ * DK * 2);
    unsigned int*   vswz   = (unsigned int*)alloc((size_t)BH * (SEQ / 32) * DK * 16 * 4);
    float*          Pws    = (float*)alloc((size_t)BH * SEQ * PREL * 4);
    unsigned short* ctxbf  = (unsigned short*)alloc(MROWS * DMODEL * 2);

    int n = (int)(MROWS * DMODEL);
    cvt_bf16_kernel<<<(n + 255) / 256, 256, 0, stream>>>(x, xbf, n);
    int nw = (int)WSWZ;
    wpack_kernel<<<(nw + 255) / 256, 256, 0, stream>>>(Wq, wq_swz);
    wpack_kernel<<<(nw + 255) / 256, 256, 0, stream>>>(Wk, wk_swz);
    wpack_kernel<<<(nw + 255) / 256, 256, 0, stream>>>(Wv, wv_swz);
    wpack_kernel<<<(nw + 255) / 256, 256, 0, stream>>>(Wo, wo_swz);
    relpack_kernel<<<(2 * PREL * 16 + 255) / 256, 256, 0, stream>>>(rel, relswz);

    dim3 gg(32, 12, 1);  // M/128, N/64
    gemm_bf16_kernel<0><<<gg, 256, 0, stream>>>(xbf, wq_swz, bq, qbf, (int)MROWS, DMODEL, DMODEL);
    gemm_bf16_kernel<0><<<gg, 256, 0, stream>>>(xbf, wk_swz, bk, kbf, (int)MROWS, DMODEL, DMODEL);
    gemm_bf16_kernel<3><<<gg, 256, 0, stream>>>(xbf, wv_swz, bv, vswz, (int)MROWS, DMODEL, DMODEL);

    pgemm_kernel<<<dim3(SEQ / 128, BH), 256, 0, stream>>>(qbf, relswz, Pws);

    attn_kernel<<<dim3(SEQ / 128, BH), 256, 0, stream>>>(qbf, kbf, vswz, Pws, ctxbf);

    gemm_bf16_kernel<2><<<gg, 256, 0, stream>>>(ctxbf, wo_swz, bo, (float*)d_out,
                                                (int)MROWS, DMODEL, DMODEL);
}
